// AbstractAttention_38302518346048
// MI455X (gfx1250) — compile-verified, matrix operand order checked by reading, not run on hardware
//
#include <hip/hip_runtime.h>

typedef __attribute__((ext_vector_type(16))) _Float16 v16h;
typedef __attribute__((ext_vector_type(8)))  _Float16 v8h;
typedef __attribute__((ext_vector_type(4)))  _Float16 v4h;
typedef __attribute__((ext_vector_type(8)))  float    v8f;
typedef __attribute__((ext_vector_type(4)))  float    v4f;
typedef __attribute__((ext_vector_type(4)))  unsigned int u32x4;
typedef __attribute__((ext_vector_type(8)))  int          i32x8;
typedef __attribute__((ext_vector_type(4)))  int          i32x4;

// TDM available? (device pass: yes; host pass: no, which conveniently selects
// the manual-copy fallback for the host parse)
#if defined(__has_builtin)
#if __has_builtin(__builtin_amdgcn_tensor_load_to_lds) && \
    __has_builtin(__builtin_amdgcn_s_wait_tensorcnt)
#define USE_TDM 1
#endif
#endif
#ifndef USE_TDM
#define USE_TDM 0
#endif

__device__ __forceinline__ v8f v8f_zero() {
  v8f z;
#pragma unroll
  for (int i = 0; i < 8; ++i) z[i] = 0.0f;
  return z;
}

__device__ __forceinline__ v8f wmma16x16x32(v16h a, v16h b, v8f c) {
  return __builtin_amdgcn_wmma_f32_16x16x32_f16(false, a, false, b, (short)0, c,
                                                false, false);
}

// 16x32 f16 fragment load, row-major source with leading dim `ld` (elements).
// CDNA5 16-bit layout: lanes 0-15 hold M=lane, K={0..7,16..23};
// lanes 16-31 hold M=lane-16, K={8..15,24..31}. B-fragments mirror with N.
__device__ __forceinline__ v16h load_frag(const _Float16* p, int ld, int row0,
                                          int k0, int lane) {
  const _Float16* q = p + (row0 + (lane & 15)) * ld + k0 + ((lane >> 4) << 3);
  v16h v;
#pragma unroll
  for (int i = 0; i < 8; ++i) {
    v[i] = q[i];
    v[i + 8] = q[i + 16];
  }
  return v;
}

#if USE_TDM
// Issue a TDM 2D f16 tile load (width x height, row stride in elements) into
// LDS with row padding (pad_interval/pad_amount per ISA 8.4: row = 2^(pi+1)
// DWORDs, then pa+1 DWORDs of pad). D# built per CDNA5 ISA §8.3/§8.4.
// Toolchain uses the 6-arg builtin: (g0, g1, g2, g3, g_extra, cpol).
__device__ __forceinline__ void tdm_load_2d(unsigned lds_off, const void* gptr,
                                            unsigned width, unsigned height,
                                            unsigned stride, unsigned pad_int,
                                            unsigned pad_amt) {
  unsigned long long ga = (unsigned long long)(size_t)gptr;
  u32x4 g0;
  g0[0] = 1u;                                   // count=1, user descriptor
  g0[1] = lds_off;                              // LDS byte address
  g0[2] = (unsigned)(ga & 0xffffffffu);         // global addr [31:0]
  g0[3] = (unsigned)((ga >> 32) & 0x01ffffffu)  // global addr [56:32]
          | (2u << 30);                         // type = 2 ("image")
  i32x8 g1;
  g1[0] = (int)((1u << 16)                      // data_size = 1 -> 2 bytes
                | (1u << 20)                    // pad_enable
                | (pad_int << 22) | (pad_amt << 25));
  g1[1] = (int)((stride & 0xffffu) << 16);      // tensor_dim0[15:0]
  g1[2] = (int)(((stride >> 16) & 0xffffu)      // tensor_dim0[31:16]
                | ((height & 0xffffu) << 16));  // tensor_dim1[15:0]
  g1[3] = (int)(((height >> 16) & 0xffffu)      // tensor_dim1[31:16]
                | ((width & 0xffffu) << 16));   // tile_dim0
  g1[4] = (int)(height & 0xffffu);              // tile_dim1 (tile_dim2 = 0)
  g1[5] = (int)stride;                          // tensor_dim0_stride[31:0]
  g1[6] = 0;
  g1[7] = 0;
  i32x4 z4 = {0, 0, 0, 0};                      // groups 2/3 unused (2D)
  i32x8 z8 = {0, 0, 0, 0, 0, 0, 0, 0};
  __builtin_amdgcn_tensor_load_to_lds(g0, g1, z4, z4, z8, 0);
}
#endif

// ---------------------------------------------------------------------------
// Pre-passes: cast activations f32->f16; transpose+cast weights so every GEMM
// becomes f16 [M,K] x [N,K]^T with plain 2D strided tiles (TDM-friendly).
// ---------------------------------------------------------------------------
__global__ __launch_bounds__(256) void cast_f16_kernel(
    const float* __restrict__ in, _Float16* __restrict__ out) {
  long long i = ((long long)blockIdx.x * 256 + threadIdx.x) * 4;
  v4f x = *(const v4f*)(in + i);
  v4h h;
#pragma unroll
  for (int j = 0; j < 4; ++j) h[j] = (_Float16)x[j];
  *(v4h*)(out + i) = h;
}

// in: f32 [nmat][rows][cols] -> out: f16 [nmat][cols][rows]
__global__ __launch_bounds__(256) void transpose_cast_kernel(
    const float* __restrict__ in, _Float16* __restrict__ out, int rows,
    int cols) {
  __shared__ float tile[32][33];
  const int m = blockIdx.z;
  const int r0 = blockIdx.y * 32, c0 = blockIdx.x * 32;
  const float* src = in + (long long)m * rows * cols;
  _Float16* dst = out + (long long)m * rows * cols;
  const int tx = threadIdx.x & 31, ty = threadIdx.x >> 5;  // 32 x 8
#pragma unroll
  for (int i = 0; i < 32; i += 8)
    tile[ty + i][tx] = src[(long long)(r0 + ty + i) * cols + c0 + tx];
  __syncthreads();
#pragma unroll
  for (int i = 0; i < 32; i += 8)
    dst[(long long)(c0 + ty + i) * rows + r0 + tx] = (_Float16)tile[tx][ty + i];
}

// ---------------------------------------------------------------------------
// GEMM: C[m,n] = sum_k A[m,k]*Bt[n,k] + bias[n].  A,Bt f16 row-major, C f32.
// Block 256 threads (8 waves), tile 128x128x32, wave tile 32x64 (2x4 WMMA).
// Double-buffered LDS; tiles fetched by the Tensor Data Mover (wave 0 issues,
// s_wait_tensorcnt + barrier hands off), manual b128 copies as fallback.
// ---------------------------------------------------------------------------
constexpr int BM = 128, BN = 128, BK = 32;
constexpr int LDT = BK + 8;  // 40 halfs: 16 DW row + 4 DW TDM pad

__global__ __launch_bounds__(256) void gemm_bias_f16_kernel(
    const _Float16* __restrict__ A, const _Float16* __restrict__ Bt,
    const float* __restrict__ bias, float* __restrict__ C, int M, int N,
    int K) {
  __shared__ _Float16 aS[2][BM * LDT];
  __shared__ _Float16 bS[2][BN * LDT];
  const int tid = threadIdx.x, lane = tid & 31, wid = tid >> 5;
  const int wm = (wid & 3) * 32, wn = (wid >> 2) * 64;
  const int n0 = blockIdx.x * BN, m0 = blockIdx.y * BM;
  const _Float16* Abase = A + (long long)m0 * K;
  const _Float16* Bbase = Bt + (long long)n0 * K;

  v8f acc[2][4];
#pragma unroll
  for (int i = 0; i < 2; ++i)
#pragma unroll
    for (int j = 0; j < 4; ++j) acc[i][j] = v8f_zero();

  auto stage = [&](int buf, int k0) {
#if USE_TDM
    if (wid == 0) {
      tdm_load_2d((unsigned)(size_t)&aS[buf][0], Abase + k0, BK, BM, K, 3, 3);
      tdm_load_2d((unsigned)(size_t)&bS[buf][0], Bbase + k0, BK, BN, K, 3, 3);
    }
#else
#pragma unroll
    for (int it = 0; it < 2; ++it) {
      int idx = tid + it * 256;  // 0..511
      int r = idx >> 2;          // 0..127
      int c = (idx & 3) << 3;    // 0,8,16,24
      *(v8h*)&aS[buf][r * LDT + c] =
          *(const v8h*)(Abase + (long long)r * K + k0 + c);
      *(v8h*)&bS[buf][r * LDT + c] =
          *(const v8h*)(Bbase + (long long)r * K + k0 + c);
    }
#endif
  };
  auto stage_handoff = [&]() {
#if USE_TDM
    if (wid == 0) __builtin_amdgcn_s_wait_tensorcnt((short)0);
#endif
    __syncthreads();
  };

  const int NK = K / BK;
  stage(0, 0);
  stage_handoff();
  for (int it = 0; it < NK; ++it) {
    const int cur = it & 1;
    if (it + 1 < NK) stage(cur ^ 1, (it + 1) * BK);  // overlap copy w/ compute
    v16h af[2], bf[4];
#pragma unroll
    for (int i = 0; i < 2; ++i)
      af[i] = load_frag(aS[cur], LDT, wm + i * 16, 0, lane);
#pragma unroll
    for (int j = 0; j < 4; ++j)
      bf[j] = load_frag(bS[cur], LDT, wn + j * 16, 0, lane);
#pragma unroll
    for (int i = 0; i < 2; ++i)
#pragma unroll
      for (int j = 0; j < 4; ++j)
        acc[i][j] = wmma16x16x32(af[i], bf[j], acc[i][j]);
    stage_handoff();
  }

  const int half = lane >> 4;
  const int nl = lane & 15;
#pragma unroll
  for (int i = 0; i < 2; ++i)
#pragma unroll
    for (int j = 0; j < 4; ++j) {
      int nc = n0 + wn + j * 16 + nl;
      float bv = bias[nc];
#pragma unroll
      for (int r = 0; r < 8; ++r) {
        int mr = m0 + wm + i * 16 + r + half * 8;
        C[(long long)mr * N + nc] = acc[i][j][r] + bv;
      }
    }
}

// ---------------------------------------------------------------------------
// RMS-norm + RoPE + cast-to-f16, permuting [B,S,H,F] -> [B,H,S,F].
// One wave per (b,s,h) vector; lane owns 4 of the 128 features.
// ---------------------------------------------------------------------------
__global__ __launch_bounds__(256) void norm_rope_cast_kernel(
    const float* __restrict__ qf, const float* __restrict__ kf,
    const float* __restrict__ vf, const float* __restrict__ qw,
    const float* __restrict__ kw, _Float16* __restrict__ qh,
    _Float16* __restrict__ kh, _Float16* __restrict__ vh) {
  constexpr int S = 2048, H = 16, F = 128;
  const int lane = threadIdx.x & 31;
  const int wid = threadIdx.x >> 5;
  const int vid = blockIdx.x * 8 + wid;  // 0 .. B*S*H-1
  const int b = vid / (S * H);
  const int rem = vid - b * (S * H);
  const int s = rem / H;
  const int h = rem - s * H;
  const long long inoff = ((long long)(b * S + s) * H + h) * F;
  const long long outoff = ((long long)(b * H + h) * S + s) * F;
  const int f0 = lane * 4;

  float cosv[4], sinv[4];
#pragma unroll
  for (int i = 0; i < 4; ++i) {
    int j = (f0 + i) & 63;
    float freq = __powf(10000.0f, -(float)j * (1.0f / 64.0f));
    float ang = (float)s * freq;
    cosv[i] = __cosf(ang);
    sinv[i] = __sinf(ang);
  }
  const float sgn = (lane < 16) ? -1.0f : 1.0f;  // rotate_half sign

  const float* src[2] = {qf, kf};
  const float* wsc[2] = {qw, kw};
  _Float16* dst[2] = {qh, kh};
#pragma unroll
  for (int t = 0; t < 2; ++t) {
    v4f x = *(const v4f*)(src[t] + inoff + f0);
    float ss = x[0] * x[0] + x[1] * x[1] + x[2] * x[2] + x[3] * x[3];
#pragma unroll
    for (int m = 1; m < 32; m <<= 1) ss += __shfl_xor(ss, m, 32);
    float rms = sqrtf(ss * (1.0f / F));
    float inv = 1.0f / (rms + 1e-6f);
    float y[4];
#pragma unroll
    for (int i = 0; i < 4; ++i) y[i] = x[i] * inv * wsc[t][f0 + i];
    float p[4];  // partner features f +/- 64 live in lane ^ 16
#pragma unroll
    for (int i = 0; i < 4; ++i) p[i] = __shfl_xor(y[i], 16, 32);
    v4h out;
#pragma unroll
    for (int i = 0; i < 4; ++i)
      out[i] = (_Float16)(y[i] * cosv[i] + sgn * p[i] * sinv[i]);
    *(v4h*)(dst[t] + outoff + f0) = out;
  }
  v4f xv = *(const v4f*)(vf + inoff + f0);
  v4h ov;
#pragma unroll
  for (int i = 0; i < 4; ++i) ov[i] = (_Float16)xv[i];
  *(v4h*)(vh + outoff + f0) = ov;
}

// ---------------------------------------------------------------------------
// Flash attention. Block = 128 threads (4 waves); wave owns 16 query rows x
// F=128. K tile staged via TDM (contiguous 64x128 f16 + LDS pad), V staged
// transposed manually (TDM cannot transpose). Online softmax stats indexed
// like the C-fragment rows (r + 8*half) so no cross-half traffic.
// ---------------------------------------------------------------------------
__global__ __launch_bounds__(128) void flash_attn_kernel(
    const _Float16* __restrict__ qh, const _Float16* __restrict__ kh,
    const _Float16* __restrict__ vh, _Float16* __restrict__ attn) {
  constexpr int S = 2048, H = 16, F = 128;
  constexpr int QT = 64, KT = 64;
  constexpr int LDK = F + 8;   // 136: 64 DW row + 4 DW TDM pad
  constexpr int LDV = KT + 8;  // 72
  __shared__ _Float16 Kt[KT * LDK];
  __shared__ _Float16 Vt[F * LDV];
  __shared__ _Float16 Pb[4][16 * LDV];

  const int tid = threadIdx.x;
  const int lane = tid & 31;
  const int wid = tid >> 5;
  const int half = lane >> 4;
  const int b = blockIdx.z, h = blockIdx.y;
  const int q0 = blockIdx.x * QT + wid * 16;
  const long long base = ((long long)(b * H + h)) * S * F;
  const _Float16* Q = qh + base;
  const _Float16* Kp = kh + base;
  const _Float16* Vp = vh + base;

  v16h qfrag[4];
#pragma unroll
  for (int kc = 0; kc < 4; ++kc) qfrag[kc] = load_frag(Q, F, q0, kc * 32, lane);

  v8f o[8];
#pragma unroll
  for (int j = 0; j < 8; ++j) o[j] = v8f_zero();
  float mst[8], lst[8];
#pragma unroll
  for (int r = 0; r < 8; ++r) {
    mst[r] = -1e30f;
    lst[r] = 0.0f;
  }
  const float scale = 0.08838834764831845f;  // 1/sqrt(128)

  for (int kt0 = 0; kt0 < S; kt0 += KT) {
    __syncthreads();  // previous tile fully consumed
#if USE_TDM
    if (wid == 0)
      tdm_load_2d((unsigned)(size_t)&Kt[0], Kp + (long long)kt0 * F, F, KT, F,
                  5, 3);
#else
#pragma unroll
    for (int it = 0; it < 8; ++it) {
      int idx = tid + it * 128;
      int kr = idx >> 4;
      int c = (idx & 15) << 3;
      *(v8h*)&Kt[kr * LDK + c] =
          *(const v8h*)(Kp + (long long)(kt0 + kr) * F + c);
    }
#endif
    // V tile transposed: Vt[f][key]
#pragma unroll
    for (int it = 0; it < 16; ++it) {
      int idx = tid + it * 128;
      int kr = idx >> 5;
      int f4 = (idx & 31) << 2;
      v4h v = *(const v4h*)(Vp + (long long)(kt0 + kr) * F + f4);
#pragma unroll
      for (int j = 0; j < 4; ++j) Vt[(f4 + j) * LDV + kr] = v[j];
    }
#if USE_TDM
    if (wid == 0) __builtin_amdgcn_s_wait_tensorcnt((short)0);
#endif
    __syncthreads();

    // Scores: 16 x KT in 4 n-tiles, K-reduction over F
    v8f sc[4];
#pragma unroll
    for (int jn = 0; jn < 4; ++jn) {
      sc[jn] = v8f_zero();
#pragma unroll
      for (int kc = 0; kc < 4; ++kc) {
        v16h kf = load_frag(Kt, LDK, jn * 16, kc * 32, lane);
        sc[jn] = wmma16x16x32(qfrag[kc], kf, sc[jn]);
      }
    }

    float alpha[8];
#pragma unroll
    for (int r = 0; r < 8; ++r) {
      float mx = sc[0][r] * scale;
#pragma unroll
      for (int jn = 1; jn < 4; ++jn) mx = fmaxf(mx, sc[jn][r] * scale);
#pragma unroll
      for (int m = 1; m < 16; m <<= 1) mx = fmaxf(mx, __shfl_xor(mx, m, 32));
      float mnew = fmaxf(mst[r], mx);
      alpha[r] = __expf(mst[r] - mnew);
      float s = 0.0f;
#pragma unroll
      for (int jn = 0; jn < 4; ++jn) {
        float p = __expf(sc[jn][r] * scale - mnew);
        s += p;
        Pb[wid][(r + half * 8) * LDV + jn * 16 + (lane & 15)] = (_Float16)p;
      }
#pragma unroll
      for (int m = 1; m < 16; m <<= 1) s += __shfl_xor(s, m, 32);
      lst[r] = lst[r] * alpha[r] + s;
      mst[r] = mnew;
    }
#pragma unroll
    for (int j = 0; j < 8; ++j)
#pragma unroll
      for (int r = 0; r < 8; ++r) o[j][r] *= alpha[r];

    asm volatile("s_wait_dscnt 0" ::: "memory");  // P stores -> P frag reads

    v16h pa[2];
#pragma unroll
    for (int kc = 0; kc < 2; ++kc)
      pa[kc] = load_frag(Pb[wid], LDV, 0, kc * 32, lane);
#pragma unroll
    for (int jf = 0; jf < 8; ++jf)
#pragma unroll
      for (int kc = 0; kc < 2; ++kc) {
        v16h vf = load_frag(Vt, LDV, jf * 16, kc * 32, lane);
        o[jf] = wmma16x16x32(pa[kc], vf, o[jf]);
      }
  }

  // Normalize by row sums and store f16 attn output as [B,S,H,F]
#pragma unroll
  for (int r = 0; r < 8; ++r) {
    float inv = 1.0f / lst[r];
    int srow = q0 + r + half * 8;
    long long outbase = ((long long)(b * S + srow)) * (H * F) + h * F;
#pragma unroll
    for (int jf = 0; jf < 8; ++jf)
      attn[outbase + jf * 16 + (lane & 15)] = (_Float16)(o[jf][r] * inv);
  }
}

// ---------------------------------------------------------------------------
extern "C" void kernel_launch(void* const* d_in, const int* in_sizes, int n_in,
                              void* d_out, int out_size, void* d_ws,
                              size_t ws_size, hipStream_t stream) {
  constexpr int Bb = 2, S = 2048, D = 2048, H = 16, F = 128;
  constexpr long long MR = (long long)Bb * S;  // 4096 rows
  constexpr long long EL = MR * D;             // 8388608 elems (act tensors)
  constexpr long long WE = (long long)H * D * F;  // 4194304 elems (weights)

  const float* xq = (const float*)d_in[0];
  const float* xk = (const float*)d_in[1];
  const float* xv = (const float*)d_in[2];
  const float* WQ = (const float*)d_in[3];
  const float* WK = (const float*)d_in[4];
  const float* WV = (const float*)d_in[5];
  const float* WO = (const float*)d_in[6];
  const float* bQ = (const float*)d_in[7];
  const float* bK = (const float*)d_in[8];
  const float* bV = (const float*)d_in[9];
  const float* bO = (const float*)d_in[10];
  const float* qw = (const float*)d_in[11];
  const float* kw = (const float*)d_in[12];

  // Workspace (~185 MB):
  //   qf,kf,vf : f32 [4096,2048] raw projections
  //   Xq,Xk,Xv : f16 [4096,2048] cast inputs   (later aliased by qh,kh,vh)
  //   WQt..WOt : f16 [2048,2048] transposed weights (Bt = [N][K])
  //   at       : f16 [B,S,H,F]  attention out  (aliases dead WQt+WKt)
  float* qf = (float*)d_ws;
  float* kf = qf + EL;
  float* vf = kf + EL;
  _Float16* Xq = (_Float16*)(vf + EL);
  _Float16* Xk = Xq + EL;
  _Float16* Xv = Xk + EL;
  _Float16* WQt = Xv + EL;
  _Float16* WKt = WQt + WE;
  _Float16* WVt = WKt + WE;
  _Float16* WOt = WVt + WE;
  _Float16* qhp = Xq;  // aliases: X* dead after projection GEMMs
  _Float16* khp = Xk;
  _Float16* vhp = Xv;
  _Float16* at = WQt;  // aliases: WQt/WKt dead after projection GEMMs

  // 1) cast activations f32 -> f16
  cast_f16_kernel<<<(int)(EL / 4 / 256), 256, 0, stream>>>(xq, Xq);
  cast_f16_kernel<<<(int)(EL / 4 / 256), 256, 0, stream>>>(xk, Xk);
  cast_f16_kernel<<<(int)(EL / 4 / 256), 256, 0, stream>>>(xv, Xv);
  // 2) transpose+cast weights: W[H,D,F] -> [H,F,D]; W_O[2048,2048] -> T
  transpose_cast_kernel<<<dim3(F / 32, D / 32, H), 256, 0, stream>>>(WQ, WQt, D,
                                                                     F);
  transpose_cast_kernel<<<dim3(F / 32, D / 32, H), 256, 0, stream>>>(WK, WKt, D,
                                                                     F);
  transpose_cast_kernel<<<dim3(F / 32, D / 32, H), 256, 0, stream>>>(WV, WVt, D,
                                                                     F);
  transpose_cast_kernel<<<dim3(D / 32, D / 32, 1), 256, 0, stream>>>(WO, WOt, D,
                                                                     D);
  // 3) Q/K/V projections (f16 WMMA, TDM-staged, f32 accum out)
  dim3 gg(D / BN, (int)(MR / BM));  // (16, 32)
  gemm_bias_f16_kernel<<<gg, 256, 0, stream>>>(Xq, WQt, bQ, qf, (int)MR, D, D);
  gemm_bias_f16_kernel<<<gg, 256, 0, stream>>>(Xk, WKt, bK, kf, (int)MR, D, D);
  gemm_bias_f16_kernel<<<gg, 256, 0, stream>>>(Xv, WVt, bV, vf, (int)MR, D, D);
  // 4) RMS-norm + RoPE + cast, permute to [B,H,S,F]
  norm_rope_cast_kernel<<<(Bb * S * H) / 8, 256, 0, stream>>>(
      qf, kf, vf, qw, kw, qhp, khp, vhp);
  // 5) flash attention -> [B,S,H,F] f16
  flash_attn_kernel<<<dim3(S / 64, H, Bb), 128, 0, stream>>>(qhp, khp, vhp, at);
  // 6) output projection + b_O -> d_out f32
  gemm_bias_f16_kernel<<<gg, 256, 0, stream>>>(at, WOt, bO, (float*)d_out,
                                               (int)MR, D, D);
}
